// GLoRI_45440753992039
// MI455X (gfx1250) — matrix operand order repaired and checked by reading, not compiled, not run on hardware
//
#include <hip/hip_runtime.h>
#include <cstdint>
#include <cstddef>

// ---------------------------------------------------------------------------
// Problem constants (match reference)
// ---------------------------------------------------------------------------
constexpr int Bc = 8, Nc = 1369, Fc = 4096, Dc = 768, Qc = 1024, Hc = 8, DFc = 2048;
constexpr int DH = Dc / Hc;            // 96
constexpr int NPAD = 1376;             // N padded to multiple of 32 for vT
constexpr float LN_EPS = 1e-5f;

typedef __bf16 bf16;
typedef __attribute__((ext_vector_type(16))) __bf16 v16bf;
typedef __attribute__((ext_vector_type(8)))  __bf16 v8bf;
typedef __attribute__((ext_vector_type(8)))  float  v8f;

#define SHUF16(lo, hi) __builtin_shufflevector(lo, hi, 0,1,2,3,4,5,6,7,8,9,10,11,12,13,14,15)

// ---------------------------------------------------------------------------
// fp32 -> bf16 convert (grid-stride)
// ---------------------------------------------------------------------------
__global__ void k_cvt_bf16(const float* __restrict__ x, bf16* __restrict__ y, size_t n)
{
    size_t i  = (size_t)blockIdx.x * blockDim.x + threadIdx.x;
    size_t st = (size_t)gridDim.x * blockDim.x;
    for (; i < n; i += st) y[i] = (bf16)x[i];
}

// ---------------------------------------------------------------------------
// bf16 GEMM:  C[M,N] = act( A[M,K] @ W[N,K]^T + bias[N] )
// A row-major [M,K], W row-major [N,K] (i.e. column-major KxN -> WMMA B).
// One wave computes a 32x64 tile: 2 A-fragments amortize 4 B-fragments over
// 8 v_wmma_f32_16x16x32_bf16 per K-step (1.5 b128 loads per WMMA).
// 8 waves / block -> block tile 256M x 64N.
// ---------------------------------------------------------------------------
__global__ void k_gemm_bt(const bf16* __restrict__ A, const bf16* __restrict__ W,
                          const float* __restrict__ bias,
                          float* __restrict__ Cf, bf16* __restrict__ Cb,
                          int M, int N, int K, int relu)
{
    const int lane = threadIdx.x & 31;
    const int wave = threadIdx.x >> 5;
    const int m0   = (blockIdx.x * 8 + wave) * 32;
    if (m0 >= M) return;                      // wave-uniform: EXEC stays all-ones
    const int n0   = blockIdx.y * 64;
    const int gh   = lane >> 4;               // half-wave id
    const int l16  = lane & 15;
    const int akoff = gh * 8;                 // A: K sub-offset per ISA layout
    const int bkoff = gh * 8;                 // B: K sub-offset per ISA layout

    int ar0 = m0 + l16;      if (ar0 >= M) ar0 = M - 1;   // clamp, store-guarded
    int ar1 = m0 + 16 + l16; if (ar1 >= M) ar1 = M - 1;
    const bf16* ap0 = A + (size_t)ar0 * K;
    const bf16* ap1 = A + (size_t)ar1 * K;

    const bf16* wp[4];
    float bi[4];
    #pragma unroll
    for (int t = 0; t < 4; ++t) {
        int col = n0 + t * 16 + l16;
        wp[t] = W + (size_t)col * K + bkoff;
        bi[t] = bias ? bias[col] : 0.f;
    }

    v8f acc0[4] = {}, acc1[4] = {};
    for (int k0 = 0; k0 < K; k0 += 32) {
        v8bf a0lo = *(const v8bf*)(ap0 + k0 + akoff);
        v8bf a0hi = *(const v8bf*)(ap0 + k0 + 16 + akoff);
        v8bf a1lo = *(const v8bf*)(ap1 + k0 + akoff);
        v8bf a1hi = *(const v8bf*)(ap1 + k0 + 16 + akoff);
        v16bf a0 = SHUF16(a0lo, a0hi);
        v16bf a1 = SHUF16(a1lo, a1hi);
        __builtin_prefetch(ap0 + k0 + 128, 0, 1);   // global_prefetch next A lines
        __builtin_prefetch(ap1 + k0 + 128, 0, 1);
        #pragma unroll
        for (int t = 0; t < 4; ++t) {
            v16bf bw = SHUF16(*(const v8bf*)(wp[t] + k0), *(const v8bf*)(wp[t] + k0 + 16));
            acc0[t] = __builtin_amdgcn_wmma_f32_16x16x32_bf16(
                          false, a0, false, bw, (short)0, acc0[t], false, false);
            acc1[t] = __builtin_amdgcn_wmma_f32_16x16x32_bf16(
                          false, a1, false, bw, (short)0, acc1[t], false, false);
        }
    }

    #pragma unroll
    for (int t = 0; t < 4; ++t) {
        int col = n0 + t * 16 + l16;
        #pragma unroll
        for (int i = 0; i < 8; ++i) {
            int row = m0 + i + gh * 8;        // C layout: VGPR i -> row i (+8 for hi half)
            if (row < M) {
                float x = acc0[t][i] + bi[t];
                if (relu) x = fmaxf(x, 0.f);
                if (Cf) Cf[(size_t)row * N + col] = x;
                if (Cb) Cb[(size_t)row * N + col] = (bf16)x;
            }
            int row1 = row + 16;
            if (row1 < M) {
                float x = acc1[t][i] + bi[t];
                if (relu) x = fmaxf(x, 0.f);
                if (Cf) Cf[(size_t)row1 * N + col] = x;
                if (Cb) Cb[(size_t)row1 * N + col] = (bf16)x;
            }
        }
    }
}

// ---------------------------------------------------------------------------
// LayerNorm helpers / kernels (256 threads, 768 features, 3 per thread)
// ---------------------------------------------------------------------------
__device__ __forceinline__ float blk_sum(float v, float* sh)
{
    #pragma unroll
    for (int m = 16; m > 0; m >>= 1) v += __shfl_xor(v, m, 32);
    int w = threadIdx.x >> 5, l = threadIdx.x & 31;
    __syncthreads();
    if (l == 0) sh[w] = v;
    __syncthreads();
    float r = 0.f;
    #pragma unroll
    for (int i = 0; i < 8; ++i) r += sh[i];
    return r;
}

// tgt0 = LN(2*query_embed) -> f32 + bf16 (batch-invariant)
__global__ void k_ln1(const float* __restrict__ qe,
                      const float* __restrict__ w, const float* __restrict__ bb,
                      float* __restrict__ outf, bf16* __restrict__ outb)
{
    __shared__ float sh[8];
    int q = blockIdx.x, t = threadIdx.x;
    float x[3];
    #pragma unroll
    for (int j = 0; j < 3; ++j) x[j] = 2.f * qe[(size_t)q * Dc + t + j * 256];
    float s = blk_sum(x[0] + x[1] + x[2], sh);
    float mean = s * (1.f / Dc);
    float vs = 0.f;
    #pragma unroll
    for (int j = 0; j < 3; ++j) { float d = x[j] - mean; vs += d * d; }
    vs = blk_sum(vs, sh);
    float rstd = rsqrtf(vs * (1.f / Dc) + LN_EPS);
    #pragma unroll
    for (int j = 0; j < 3; ++j) {
        int c = t + j * 256;
        float y = (x[j] - mean) * rstd * w[c] + bb[c];
        outf[(size_t)q * Dc + c] = y;
        outb[(size_t)q * Dc + c] = (bf16)y;
    }
}

// tgt2 = LN(tgt0[q] + oproj[b,q]) -> f32 + bf16
__global__ void k_ln2(const float* __restrict__ tgt0, const float* __restrict__ opj,
                      const float* __restrict__ w, const float* __restrict__ bb,
                      float* __restrict__ outf, bf16* __restrict__ outb)
{
    __shared__ float sh[8];
    int r = blockIdx.x, q = r & (Qc - 1), t = threadIdx.x;
    float x[3];
    #pragma unroll
    for (int j = 0; j < 3; ++j) {
        int c = t + j * 256;
        x[j] = tgt0[(size_t)q * Dc + c] + opj[(size_t)r * Dc + c];
    }
    float mean = blk_sum(x[0] + x[1] + x[2], sh) * (1.f / Dc);
    float vs = 0.f;
    #pragma unroll
    for (int j = 0; j < 3; ++j) { float d = x[j] - mean; vs += d * d; }
    float rstd = rsqrtf(blk_sum(vs, sh) * (1.f / Dc) + LN_EPS);
    #pragma unroll
    for (int j = 0; j < 3; ++j) {
        int c = t + j * 256;
        float y = (x[j] - mean) * rstd * w[c] + bb[c];
        outf[(size_t)r * Dc + c] = y;
        outb[(size_t)r * Dc + c] = (bf16)y;
    }
}

// LN3 + concat([cls, h]) . dup_pool[q] + dup_bias[q]  -> logits[b,q]
__global__ void k_final(const float* __restrict__ tgt2, const float* __restrict__ ff2,
                        const float* __restrict__ cls,
                        const float* __restrict__ n3w, const float* __restrict__ n3b,
                        const float* __restrict__ dpool, const float* __restrict__ dbias,
                        float* __restrict__ out)
{
    __shared__ float sh[8];
    int r = blockIdx.x, b = r >> 10, q = r & (Qc - 1), t = threadIdx.x;
    float x[3];
    #pragma unroll
    for (int j = 0; j < 3; ++j) {
        int c = t + j * 256;
        x[j] = tgt2[(size_t)r * Dc + c] + ff2[(size_t)r * Dc + c];
    }
    float mean = blk_sum(x[0] + x[1] + x[2], sh) * (1.f / Dc);
    float vs = 0.f;
    #pragma unroll
    for (int j = 0; j < 3; ++j) { float d = x[j] - mean; vs += d * d; }
    float rstd = rsqrtf(blk_sum(vs, sh) * (1.f / Dc) + LN_EPS);

    const float* dp = dpool + (size_t)q * (Fc + Dc);
    float dot = 0.f;
    #pragma unroll
    for (int j = 0; j < 3; ++j) {
        int c = t + j * 256;
        float hv = (x[j] - mean) * rstd * n3w[c] + n3b[c];
        dot += hv * dp[Fc + c];                       // h part: offset F in concat
    }
    for (int jj = t; jj < Fc; jj += 256)              // cls part
        dot += cls[(size_t)b * Fc + jj] * dp[jj];
    dot = blk_sum(dot, sh);
    if (t == 0) out[r] = dot + dbias[q];
}

// ---------------------------------------------------------------------------
// V transpose: vT[b,h,d,n] (bf16, n padded to NPAD, zero tail) from vh[b*n, D]
// ---------------------------------------------------------------------------
__global__ void k_transpose_v(const bf16* __restrict__ vh, bf16* __restrict__ vT)
{
    int n = blockIdx.x * 256 + threadIdx.x;
    if (n >= NPAD) return;
    int d = blockIdx.y, bh = blockIdx.z;
    int b = bh >> 3, h = bh & 7;
    bf16 v = (bf16)0.f;
    if (n < Nc) v = vh[((size_t)b * Nc + n) * Dc + h * DH + d];
    vT[((size_t)bh * DH + d) * NPAD + n] = v;
}

// ---------------------------------------------------------------------------
// Flash attention: one wave per (b, h, 16-row q tile). 32-key tiles.
// S = Q K^T / sqrt(96) via 2x3 WMMAs, online softmax, O += P V via 6 WMMAs.
// P reshaped C-layout -> A-layout through a 1KB LDS bounce (in-order DS ops).
// ---------------------------------------------------------------------------
__global__ void k_attn(const bf16* __restrict__ qh, const bf16* __restrict__ kh,
                       const bf16* __restrict__ vT, bf16* __restrict__ oat)
{
    const int q0 = blockIdx.x * 16;
    const int h  = blockIdx.y;
    const int b  = blockIdx.z;
    const int lane = threadIdx.x;
    const int gh = lane >> 4, l16 = lane & 15;
    const int akoff = gh * 8, bkoff = gh * 8;
    const float scale = 0.1020620726159658f;       // 1/sqrt(96)

    __shared__ __align__(32) bf16 Pl[16 * 32];

    // Q tile A-fragments (3 K-chunks of 32 over head dim 96); q is batch-invariant
    v16bf aq[3];
    {
        const bf16* qp = qh + (size_t)(q0 + l16) * Dc + h * DH;
        #pragma unroll
        for (int c = 0; c < 3; ++c) {
            v8bf lo = *(const v8bf*)(qp + c * 32 + akoff);
            v8bf hi = *(const v8bf*)(qp + c * 32 + 16 + akoff);
            aq[c] = SHUF16(lo, hi);
        }
    }

    v8f acc[6] = {};
    float mrow[8], lrow[8];
    #pragma unroll
    for (int i = 0; i < 8; ++i) { mrow[i] = -3.0e38f; lrow[i] = 0.f; }

    const bf16* vTb = vT + ((size_t)(b * Hc + h) * DH) * NPAD;

    for (int n0 = 0; n0 < Nc; n0 += 32) {
        // --- scores: two 16x16 tiles covering 32 keys ---
        v8f s[2];
        #pragma unroll
        for (int t = 0; t < 2; ++t) {
            int key  = n0 + t * 16 + l16;                 // B column = key index
            int keyc = key < Nc ? key : Nc - 1;
            const bf16* kp = kh + ((size_t)b * Nc + keyc) * Dc + h * DH + bkoff;
            v8f sv = {};
            #pragma unroll
            for (int c = 0; c < 3; ++c) {
                v16bf bk = SHUF16(*(const v8bf*)(kp + c * 32), *(const v8bf*)(kp + c * 32 + 16));  // table-order K: K-cache rows are B cols
                sv = __builtin_amdgcn_wmma_f32_16x16x32_bf16(
                         false, aq[c], false, bk, (short)0, sv, false, false);
            }
            #pragma unroll
            for (int i = 0; i < 8; ++i)
                sv[i] = (key < Nc) ? sv[i] * scale : -3.0e38f;   // mask pad keys
            s[t] = sv;
        }
        // --- online softmax stats (row i lives across 16 lanes of one half) ---
        float corr[8];
        #pragma unroll
        for (int i = 0; i < 8; ++i) {
            float tm = fmaxf(s[0][i], s[1][i]);
            for (int m = 8; m > 0; m >>= 1) tm = fmaxf(tm, __shfl_xor(tm, m, 32));
            float mnew = fmaxf(mrow[i], tm);
            corr[i] = __expf(mrow[i] - mnew);
            mrow[i] = mnew;
        }
        __syncthreads();
        #pragma unroll
        for (int i = 0; i < 8; ++i) {
            float p0 = __expf(s[0][i] - mrow[i]);
            float p1 = __expf(s[1][i] - mrow[i]);
            int row = i + gh * 8;
            Pl[row * 32 + l16]      = (bf16)p0;
            Pl[row * 32 + 16 + l16] = (bf16)p1;
            float ps = p0 + p1;
            for (int m = 8; m > 0; m >>= 1) ps += __shfl_xor(ps, m, 32);
            lrow[i] = lrow[i] * corr[i] + ps;
        }
        __syncthreads();
        // --- reload P as A-fragment, accumulate O += P V ---
        v8bf plo = *(const v8bf*)(&Pl[l16 * 32 + akoff]);
        v8bf phi = *(const v8bf*)(&Pl[l16 * 32 + 16 + akoff]);
        v16bf pa = SHUF16(plo, phi);
        #pragma unroll
        for (int td = 0; td < 6; ++td) {
            v8f o = acc[td];
            #pragma unroll
            for (int i = 0; i < 8; ++i) o[i] *= corr[i];
            v16bf bv = SHUF16(*(const v8bf*)(vTb + (size_t)(td * 16 + l16) * NPAD + n0 + bkoff), *(const v8bf*)(vTb + (size_t)(td * 16 + l16) * NPAD + n0 + 16 + bkoff));
            acc[td] = __builtin_amdgcn_wmma_f32_16x16x32_bf16(
                          false, pa, false, bv, (short)0, o, false, false);
        }
    }
    // --- normalize and store bf16 (feeds out_proj GEMM) ---
    #pragma unroll
    for (int td = 0; td < 6; ++td)
        #pragma unroll
        for (int i = 0; i < 8; ++i) {
            float v = acc[td][i] / lrow[i];
            int row = q0 + i + gh * 8;
            int col = h * DH + td * 16 + l16;
            oat[((size_t)b * Qc + row) * Dc + col] = (bf16)v;
        }
}

// ---------------------------------------------------------------------------
// Host-side pipeline
// ---------------------------------------------------------------------------
extern "C" void kernel_launch(void* const* d_in, const int* in_sizes, int n_in,
                              void* d_out, int out_size, void* d_ws, size_t ws_size,
                              hipStream_t stream)
{
    const float* pt  = (const float*)d_in[0];   // patch_tokens [B,N,F]
    const float* cls = (const float*)d_in[1];   // cls_tok      [B,F]
    const float* ew  = (const float*)d_in[2];   // embed_w      [D,F]
    const float* eb  = (const float*)d_in[3];   // embed_b      [D]
    const float* qe  = (const float*)d_in[4];   // query_embed  [Q,D]
    const float* ipw = (const float*)d_in[5];   // in_proj_w    [3D,D]
    const float* ipb = (const float*)d_in[6];   // in_proj_b    [3D]
    const float* opw = (const float*)d_in[7];   // out_proj_w   [D,D]
    const float* opb = (const float*)d_in[8];   // out_proj_b   [D]
    const float* l1w = (const float*)d_in[9];   // lin1_w       [DF,D]
    const float* l1b = (const float*)d_in[10];
    const float* l2w = (const float*)d_in[11];  // lin2_w       [D,DF]
    const float* l2b = (const float*)d_in[12];
    const float* n1w = (const float*)d_in[13];
    const float* n1b = (const float*)d_in[14];
    const float* n2w = (const float*)d_in[15];
    const float* n2b = (const float*)d_in[16];
    const float* n3w = (const float*)d_in[17];
    const float* n3b = (const float*)d_in[18];
    const float* dpool = (const float*)d_in[19];
    const float* dbias = (const float*)d_in[20];
    float* out = (float*)d_out;

    const size_t BN = (size_t)Bc * Nc;          // 10952
    const size_t BQ = (size_t)Bc * Qc;          // 8192

    // --- carve workspace ---
    char* p = (char*)d_ws;
    auto carve = [&](size_t bytes) -> void* {
        void* r = (void*)p; p += (bytes + 255) & ~(size_t)255; return r;
    };
    bf16*  pt_bf   = (bf16*) carve(BN * Fc * sizeof(bf16));
    bf16*  ew_bf   = (bf16*) carve((size_t)Dc * Fc * sizeof(bf16));
    bf16*  ipw_bf  = (bf16*) carve((size_t)3 * Dc * Dc * sizeof(bf16));
    bf16*  opw_bf  = (bf16*) carve((size_t)Dc * Dc * sizeof(bf16));
    bf16*  l1w_bf  = (bf16*) carve((size_t)DFc * Dc * sizeof(bf16));
    bf16*  l2w_bf  = (bf16*) carve((size_t)Dc * DFc * sizeof(bf16));
    float* tgt0_f  = (float*)carve((size_t)Qc * Dc * sizeof(float));
    bf16*  tgt0_bf = (bf16*) carve((size_t)Qc * Dc * sizeof(bf16));
    bf16*  mem_bf  = (bf16*) carve(BN * Dc * sizeof(bf16));
    bf16*  kh_bf   = (bf16*) carve(BN * Dc * sizeof(bf16));
    bf16*  vh_bf   = (bf16*) carve(BN * Dc * sizeof(bf16));
    bf16*  vT_bf   = (bf16*) carve((size_t)Bc * Hc * DH * NPAD * sizeof(bf16));
    bf16*  qh_bf   = (bf16*) carve((size_t)Qc * Dc * sizeof(bf16));
    bf16*  oat_bf  = (bf16*) carve(BQ * Dc * sizeof(bf16));
    float* opj_f   = (float*)carve(BQ * Dc * sizeof(float));
    float* tgt2_f  = (float*)carve(BQ * Dc * sizeof(float));
    bf16*  tgt2_bf = (bf16*) carve(BQ * Dc * sizeof(bf16));
    bf16*  ff1_bf  = (bf16*) carve(BQ * DFc * sizeof(bf16));
    float* ff2_f   = (float*)carve(BQ * Dc * sizeof(float));

    auto cvt = [&](const float* s, bf16* d, size_t n) {
        k_cvt_bf16<<<dim3(2048), dim3(256), 0, stream>>>(s, d, n);
    };
    auto gemm = [&](const bf16* A, const bf16* W, const float* bias,
                    float* Cf, bf16* Cb, int M, int N, int K, int relu) {
        dim3 g((M + 255) / 256, N / 64);
        k_gemm_bt<<<g, dim3(256), 0, stream>>>(A, W, bias, Cf, Cb, M, N, K, relu);
    };

    // 1) convert operands to bf16 (weights fit in L2; big A matrices read once)
    cvt(pt,  pt_bf,  BN * Fc);
    cvt(ew,  ew_bf,  (size_t)Dc * Fc);
    cvt(ipw, ipw_bf, (size_t)3 * Dc * Dc);
    cvt(opw, opw_bf, (size_t)Dc * Dc);
    cvt(l1w, l1w_bf, (size_t)DFc * Dc);
    cvt(l2w, l2w_bf, (size_t)Dc * DFc);

    // 2) tgt0 = LN(2*query_embed)  (batch-invariant)
    k_ln1<<<dim3(Qc), dim3(256), 0, stream>>>(qe, n1w, n1b, tgt0_f, tgt0_bf);

    // 3) mem = relu(patch @ embed_w^T + b)      [10952 x 768, K=4096]
    gemm(pt_bf, ew_bf, eb, nullptr, mem_bf, (int)BN, Dc, Fc, 1);

    // 4) K,V,Q projections
    gemm(mem_bf, ipw_bf + (size_t)Dc * Dc,     ipb + Dc,     nullptr, kh_bf, (int)BN, Dc, Dc, 0);
    gemm(mem_bf, ipw_bf + (size_t)2 * Dc * Dc, ipb + 2 * Dc, nullptr, vh_bf, (int)BN, Dc, Dc, 0);
    gemm(tgt0_bf, ipw_bf,                      ipb,          nullptr, qh_bf, Qc, Dc, Dc, 0);

    // 5) V -> [b,h,d,n] for contiguous B-fragments in P*V
    k_transpose_v<<<dim3((NPAD + 255) / 256, DH, Bc * Hc), dim3(256), 0, stream>>>(vh_bf, vT_bf);

    // 6) flash attention -> o_attn (bf16)
    k_attn<<<dim3(Qc / 16, Hc, Bc), dim3(32), 0, stream>>>(qh_bf, kh_bf, vT_bf, oat_bf);

    // 7) out_proj -> f32
    gemm(oat_bf, opw_bf, opb, opj_f, nullptr, (int)BQ, Dc, Dc, 0);

    // 8) tgt2 = LN(tgt0 + oproj)
    k_ln2<<<dim3((int)BQ), dim3(256), 0, stream>>>(tgt0_f, opj_f, n2w, n2b, tgt2_f, tgt2_bf);

    // 9) FFN
    gemm(tgt2_bf, l1w_bf, l1b, nullptr, ff1_bf, (int)BQ, DFc, Dc, 1);
    gemm(ff1_bf,  l2w_bf, l2b, ff2_f, nullptr, (int)BQ, Dc, DFc, 0);

    // 10) LN3 + concat dot -> logits
    k_final<<<dim3((int)BQ), dim3(256), 0, stream>>>(tgt2_f, ff2_f, cls, n3w, n3b,
                                                     dpool, dbias, out);
}
